// MultiHeadAttention_35777077576512
// MI455X (gfx1250) — matrix operand order repaired and checked by reading, not compiled, not run on hardware
//
#include <hip/hip_runtime.h>
#include <cstdint>
#include <cstddef>

// ---------------------------------------------------------------------------
// MI455X (gfx1250) fused multi-head attention, f16 WMMA path.
// GEMMs use double-buffered GLOBAL_LOAD_ASYNC_TO_LDS_B128 panel staging with
// DS-path fragment reads (address-space-inferred ds_load_b128).
// Requires ws_size >= 176,160,768 bytes.
// ---------------------------------------------------------------------------

typedef __attribute__((ext_vector_type(16))) _Float16 v16h;
typedef __attribute__((ext_vector_type(8)))  _Float16 v8h;
typedef __attribute__((ext_vector_type(4)))  _Float16 v4h;
typedef __attribute__((ext_vector_type(8)))  float    v8f;

#define BDIM   64
#define NDISKS 253
#define SEQ    256
#define DM     1024
#define NH     16
#define DK     64

#define APITCH 40   // halves, 128x32 A panel rows (80B rows, bank-stride 20)
#define BPITCH 48   // halves, 128x32 B panel rows (96B rows, 32B aligned)

__device__ __forceinline__ v8f wmma_f16(v16h a, v16h b, v8f c) {
  // D = A(16x32 f16) * B(32x16 f16) + C(16x16 f32)
  return __builtin_amdgcn_wmma_f32_16x16x32_f16(false, a, false, b, (short)0, c,
                                                false, false);
}

// A-matrix fragment (16x32 f16), contiguous k order used by this code: lane holds row (lane&15);
// lanes 0-15 carry K 0-15, lanes 16-31 carry K 16-31.
__device__ __forceinline__ v16h ld_frag_a(const _Float16* rowbase, int kbase, int lane) {
  const int kh = (lane & 16) ? 16 : 0;
  union { v16h v; v8h h[2]; } u;
  u.h[0] = *(const v8h*)(rowbase + kbase + kh);
  u.h[1] = *(const v8h*)(rowbase + kbase + kh + 8);
  return u.v;
}

// B-matrix fragment (32x16 f16), dense column-major B: lane holds column
// (lane&15); lanes 0-15 carry K 0-15, lanes 16-31 carry K 16-31.
__device__ __forceinline__ v16h ld_frag_b(const _Float16* colbase, int kbase, int lane) {
  const int kh = (lane & 16) ? 16 : 0;
  return *(const v16h*)(colbase + kbase + kh);
}

// One wave-wide async global->LDS b128 copy (ASYNCcnt-tracked, no VGPR data).
__device__ __forceinline__ void async_copy_b128(const _Float16* lds_dst,
                                                const _Float16* gsrc) {
  const uint32_t loff = (uint32_t)(uintptr_t)lds_dst;  // LDS aperture: low 32 bits
  const unsigned long long ga = (unsigned long long)(uintptr_t)gsrc;
  asm volatile("global_load_async_to_lds_b128 %0, %1, off"
               :: "v"(loff), "v"(ga)
               : "memory");
}

// Fill one double-buffer slot: A panel 128x32 (pitch APITCH) and B panel
// 128x32 (pitch BPITCH). 4 async instructions per wave -> ASYNCcnt += 4.
__device__ __forceinline__ void fill_panels(const _Float16* __restrict__ Ag,
                                            const _Float16* __restrict__ Bg,
                                            _Float16* A_s, _Float16* B_s,
                                            int kk, int t) {
#pragma unroll
  for (int c = 0; c < 2; ++c) {
    const int ch  = t + c * 256;       // 0..511
    const int r   = ch >> 2;           // 0..127 panel row
    const int seg = (ch & 3) * 8;      // 16B segment within 32-half row
    async_copy_b128(A_s + r * APITCH + seg, Ag + (size_t)r * DM + kk + seg);
    async_copy_b128(B_s + r * BPITCH + seg, Bg + (size_t)r * DM + kk + seg);
  }
}

// Shared pipelined mainloop: WG computes a 128x128 tile; wave (wm,wn) owns a
// 32x64 sub-tile (2x4 WMMA accumulators). Shared buffers are passed as array
// references so fragment reads keep AS(3) provenance -> ds_load_b128.
__device__ __forceinline__ void gemm_mainloop(
    const _Float16* __restrict__ Ag, const _Float16* __restrict__ Bg,
    _Float16 (&A_s)[2][128 * APITCH], _Float16 (&B_s)[2][128 * BPITCH],
    v8f acc[2][4], int t, int lane, int wm, int wn, int ln16) {
  fill_panels(Ag, Bg, &A_s[0][0], &B_s[0][0], 0, t);
  int cur = 0;
  for (int i = 0; i < DM / 32; ++i) {
    if (i + 1 < DM / 32) {
      fill_panels(Ag, Bg, &A_s[cur ^ 1][0], &B_s[cur ^ 1][0], (i + 1) * 32, t);
      asm volatile("s_wait_asynccnt 0x4" ::: "memory");  // prev 4 done (in order)
    } else {
      asm volatile("s_wait_asynccnt 0x0" ::: "memory");
    }
    __syncthreads();
    v16h af[2], bf[4];
#pragma unroll
    for (int mt = 0; mt < 2; ++mt)
      af[mt] =
          ld_frag_a(&A_s[cur][(wm * 32 + mt * 16 + ln16) * APITCH], 0, lane);
#pragma unroll
    for (int nt = 0; nt < 4; ++nt)
      bf[nt] =
          ld_frag_b(&B_s[cur][(wn * 64 + nt * 16 + ln16) * BPITCH], 0, lane);
#pragma unroll
    for (int mt = 0; mt < 2; ++mt)
#pragma unroll
      for (int nt = 0; nt < 4; ++nt)
        acc[mt][nt] = wmma_f16(af[mt], bf[nt], acc[mt][nt]);
    __syncthreads();
    cur ^= 1;
  }
}

// ---------------------------------------------------------------------------
// Kernel 1: LayerNorm(x) -> f16   (one 256-thread WG per (b,s) row)
// ---------------------------------------------------------------------------
__global__ __launch_bounds__(256) void ln_kernel(const float* __restrict__ x,
                                                 _Float16* __restrict__ xh) {
  const int row  = blockIdx.x;
  const int t    = threadIdx.x;
  const int lane = t & 31, wave = t >> 5;
  const float4 v = *(const float4*)(x + (size_t)row * DM + t * 4);
  float s1 = v.x + v.y + v.z + v.w;
  float s2 = v.x * v.x + v.y * v.y + v.z * v.z + v.w * v.w;
#pragma unroll
  for (int m = 1; m < 32; m <<= 1) {
    s1 += __shfl_xor(s1, m, 32);
    s2 += __shfl_xor(s2, m, 32);
  }
  __shared__ float r1[8], r2[8];
  if (lane == 0) { r1[wave] = s1; r2[wave] = s2; }
  __syncthreads();
  float t1 = 0.f, t2 = 0.f;
#pragma unroll
  for (int i = 0; i < 8; ++i) { t1 += r1[i]; t2 += r2[i]; }
  const float mu   = t1 * (1.0f / DM);
  const float var  = t2 * (1.0f / DM) - mu * mu;
  const float rstd = rsqrtf(var + 1e-5f);
  v4h o;
  o[0] = (_Float16)((v.x - mu) * rstd);
  o[1] = (_Float16)((v.y - mu) * rstd);
  o[2] = (_Float16)((v.z - mu) * rstd);
  o[3] = (_Float16)((v.w - mu) * rstd);
  *(v4h*)(xh + (size_t)row * DM + t * 4) = o;
}

// ---------------------------------------------------------------------------
// Kernel 2: f32 -> f16 weight convert (1M elements per launch)
// ---------------------------------------------------------------------------
__global__ __launch_bounds__(256) void cvt_kernel(const float* __restrict__ src,
                                                  _Float16* __restrict__ dst) {
  const int i = blockIdx.x * 256 + threadIdx.x;
  dst[i] = (_Float16)src[i];
}

// ---------------------------------------------------------------------------
// Kernel 3: QKV projection.  out[b,h,s,dk] = xn[b,s,:] . W[e,:],  e=h*64+dk
// Double-buffered async-LDS pipeline; Q pre-scaled by 1/sqrt(d_k).
// ---------------------------------------------------------------------------
__global__ __launch_bounds__(256) void gemm_qkv_kernel(
    const _Float16* __restrict__ xh, const _Float16* __restrict__ wq,
    const _Float16* __restrict__ wk, const _Float16* __restrict__ wv,
    _Float16* __restrict__ Qh, _Float16* __restrict__ Kh,
    _Float16* __restrict__ Vh) {
  const int b  = blockIdx.z;
  const int w  = blockIdx.y;
  const int tm = blockIdx.x & 1, tn = blockIdx.x >> 1;
  const int t = threadIdx.x;
  const int lane = t & 31, wave = t >> 5;
  const int ln16 = lane & 15, grp = lane >> 4;
  const int wm = wave & 3, wn = wave >> 2;
  const int m0 = tm * 128 + wm * 32;
  const int n0 = tn * 128 + wn * 64;

  __shared__ _Float16 A_s[2][128 * APITCH] __attribute__((aligned(64)));
  __shared__ _Float16 B_s[2][128 * BPITCH] __attribute__((aligned(64)));

  const _Float16* Ag = xh + (size_t)b * SEQ * DM + (size_t)(tm * 128) * DM;
  const _Float16* W  = (w == 0) ? wq : (w == 1) ? wk : wv;
  const _Float16* Bg = W + (size_t)(tn * 128) * DM;
  _Float16* Out      = (w == 0) ? Qh : (w == 1) ? Kh : Vh;
  const float scale  = (w == 0) ? 0.125f : 1.0f;

  v8f zero = {};
  v8f acc[2][4];
#pragma unroll
  for (int mt = 0; mt < 2; ++mt)
#pragma unroll
    for (int nt = 0; nt < 4; ++nt) acc[mt][nt] = zero;

  gemm_mainloop(Ag, Bg, A_s, B_s, acc, t, lane, wm, wn, ln16);

#pragma unroll
  for (int mt = 0; mt < 2; ++mt) {
#pragma unroll
    for (int nt = 0; nt < 4; ++nt) {
      const int e  = n0 + nt * 16 + ln16;
      const int hh = e >> 6, dk = e & 63;
#pragma unroll
      for (int r = 0; r < 8; ++r) {
        const int s = m0 + mt * 16 + r + 8 * grp;
        Out[(((size_t)(b * NH + hh)) * SEQ + s) * DK + dk] =
            (_Float16)(acc[mt][nt][r] * scale);
      }
    }
  }
}

// ---------------------------------------------------------------------------
// Kernel 4: flash attention per (b,h) with structural bias + mask.
// 8 waves x 32 query rows; online softmax; K from L2, V transposed in LDS.
// ---------------------------------------------------------------------------
__global__ __launch_bounds__(256) void attn_kernel(
    const _Float16* __restrict__ Qh, const _Float16* __restrict__ Kh,
    const _Float16* __restrict__ Vh, const int* __restrict__ state,
    const unsigned char* __restrict__ pmask, const float* __restrict__ etb,
    const float* __restrict__ srb, const float* __restrict__ rsb,
    _Float16* __restrict__ ctx) {
  const int h = blockIdx.x, b = blockIdx.y;
  const int t = threadIdx.x;
  const int lane = t & 31, wave = t >> 5;
  const int ln16 = lane & 15, grp = lane >> 4;

  __shared__ _Float16 Vt_s[DK * 272] __attribute__((aligned(64)));   // 34.8 KB
  __shared__ _Float16 p_s[8][16 * 40] __attribute__((aligned(64)));  // 10.2 KB
  __shared__ int state_s[SEQ];
  __shared__ unsigned char mask_s[SEQ];

  const size_t hoff = ((size_t)(b * NH + h)) * SEQ * DK;
  const _Float16* Qp = Qh + hoff;
  const _Float16* Kp = Kh + hoff;
  const _Float16* Vp = Vh + hoff;

  if (t < 3) { state_s[t] = -1 - t; mask_s[t] = 1; }
  else       { state_s[t] = state[b * NDISKS + (t - 3)];
               mask_s[t]  = pmask[b * NDISKS + (t - 3)]; }

  // stage V transposed: Vt_s[dk][s]
#pragma unroll
  for (int i = 0; i < 8; ++i) {
    const int idx = i * 256 + t;
    const int s = idx >> 3, j = (idx & 7) * 8;
    v8h vv = *(const v8h*)(Vp + (size_t)s * DK + j);
#pragma unroll
    for (int u = 0; u < 8; ++u) Vt_s[(j + u) * 272 + s] = vv[u];
  }
  __syncthreads();

  const float et0 = etb[0], et1 = etb[1], et2 = etb[2], et3 = etb[3];
  const float sr0 = srb[0], sr1 = srb[1];
  const float rs0 = rsb[0], rs2 = rsb[2];

  const int qb = wave * 32;
  v16h aq[2][2];
#pragma unroll
  for (int mt = 0; mt < 2; ++mt) {
    const _Float16* rp = Qp + (size_t)(qb + mt * 16 + ln16) * DK;
#pragma unroll
    for (int ks = 0; ks < 2; ++ks) aq[mt][ks] = ld_frag_a(rp, ks * 32, lane);
  }

  v8f zero = {};
  v8f oacc[2][4];
  float mrow[2][8], lrow[2][8];
#pragma unroll
  for (int mt = 0; mt < 2; ++mt) {
#pragma unroll
    for (int j = 0; j < 4; ++j) oacc[mt][j] = zero;
#pragma unroll
    for (int r = 0; r < 8; ++r) { mrow[mt][r] = -1e30f; lrow[mt][r] = 0.f; }
  }

  for (int kt = 0; kt < 8; ++kt) {
    const int kb = kt * 32;
    v16h bk[2][2];
#pragma unroll
    for (int nt = 0; nt < 2; ++nt) {
      const _Float16* cp = Kp + (size_t)(kb + nt * 16 + ln16) * DK;
#pragma unroll
      for (int ks = 0; ks < 2; ++ks) bk[nt][ks] = ld_frag_b(cp, ks * 32, lane);
    }
    v16h bv[4];
#pragma unroll
    for (int j = 0; j < 4; ++j)
      bv[j] = ld_frag_b(&Vt_s[(size_t)(j * 16 + ln16) * 272], kb, lane);

#pragma unroll
    for (int mt = 0; mt < 2; ++mt) {
      v8f sa0 = zero, sa1 = zero;
#pragma unroll
      for (int ks = 0; ks < 2; ++ks) {
        sa0 = wmma_f16(aq[mt][ks], bk[0][ks], sa0);
        sa1 = wmma_f16(aq[mt][ks], bk[1][ks], sa1);
      }
      const int q0 = qb + mt * 16 + 8 * grp;
#pragma unroll
      for (int r = 0; r < 8; ++r) {
        const int q   = q0 + r;
        const bool qok = mask_s[q] != 0;
        const bool qd  = q >= 3;
        const int  qs  = state_s[q];
        float best = -1e30f;
        float sc[2];
#pragma unroll
        for (int nt = 0; nt < 2; ++nt) {
          const int k = kb + nt * 16 + ln16;
          float v = (nt == 0) ? sa0[r] : sa1[r];
          if (q != k) {
            float eb = qd ? ((k < 3) ? et1 : et0) : ((k < 3) ? et3 : et2);
            if (qd && k >= 3)
              eb += ((qs == state_s[k]) ? sr0 : sr1) + ((q < k) ? rs0 : rs2);
            v += eb;
          }
          if (!(qok && mask_s[k])) v = -1e30f;
          sc[nt] = v;
          best = fmaxf(best, v);
        }
#pragma unroll
        for (int m = 1; m < 16; m <<= 1)
          best = fmaxf(best, __shfl_xor(best, m, 32));
        const float mo = mrow[mt][r];
        const float mn = fmaxf(mo, best);
        const float alpha = __expf(mo - mn);
        const float p0 = __expf(sc[0] - mn);
        const float p1 = __expf(sc[1] - mn);
        float ps = p0 + p1;
#pragma unroll
        for (int m = 1; m < 16; m <<= 1) ps += __shfl_xor(ps, m, 32);
        mrow[mt][r] = mn;
        lrow[mt][r] = lrow[mt][r] * alpha + ps;
#pragma unroll
        for (int j = 0; j < 4; ++j) oacc[mt][j][r] *= alpha;
        sa0[r] = p0;
        sa1[r] = p1;
      }

      // C-layout -> A-layout conversion of P through per-wave LDS scratch
#pragma unroll
      for (int r = 0; r < 8; ++r) {
        const int row = r + 8 * grp;
        p_s[wave][row * 40 + ln16]      = (_Float16)sa0[r];
        p_s[wave][row * 40 + 16 + ln16] = (_Float16)sa1[r];
      }
      __builtin_amdgcn_wave_barrier();
      asm volatile("s_wait_dscnt 0" ::: "memory");
      v16h ap = ld_frag_a(&p_s[wave][ln16 * 40], 0, lane);
      __builtin_amdgcn_wave_barrier();
#pragma unroll
      for (int j = 0; j < 4; ++j) oacc[mt][j] = wmma_f16(ap, bv[j], oacc[mt][j]);
    }
  }

  // normalize & store context (f16), masked query rows -> 0 (nan_to_num)
#pragma unroll
  for (int mt = 0; mt < 2; ++mt) {
    const int q0 = qb + mt * 16 + 8 * grp;
#pragma unroll
    for (int r = 0; r < 8; ++r) {
      const int q = q0 + r;
      const float l = lrow[mt][r];
      const float inv = (mask_s[q] && l > 0.f) ? 1.0f / l : 0.0f;
      _Float16* dst = ctx + ((size_t)(b * SEQ + q)) * DM + h * DK;
#pragma unroll
      for (int j = 0; j < 4; ++j)
        dst[j * 16 + ln16] = (_Float16)(oacc[mt][j][r] * inv);
    }
  }
}

// ---------------------------------------------------------------------------
// Kernel 5: output projection + bias + residual (f32 out), async-LDS pipeline.
// ---------------------------------------------------------------------------
__global__ __launch_bounds__(256) void gemm_out_kernel(
    const _Float16* __restrict__ ctx, const _Float16* __restrict__ wo,
    const float* __restrict__ x, const float* __restrict__ bO,
    float* __restrict__ out) {
  const int b  = blockIdx.z;
  const int tm = blockIdx.x & 1, tn = blockIdx.x >> 1;
  const int t = threadIdx.x;
  const int lane = t & 31, wave = t >> 5;
  const int ln16 = lane & 15, grp = lane >> 4;
  const int wm = wave & 3, wn = wave >> 2;
  const int m0 = tm * 128 + wm * 32;
  const int n0 = tn * 128 + wn * 64;

  __shared__ _Float16 A_s[2][128 * APITCH] __attribute__((aligned(64)));
  __shared__ _Float16 B_s[2][128 * BPITCH] __attribute__((aligned(64)));

  const _Float16* Ag = ctx + (size_t)b * SEQ * DM + (size_t)(tm * 128) * DM;
  const _Float16* Bg = wo + (size_t)(tn * 128) * DM;

  v8f zero = {};
  v8f acc[2][4];
#pragma unroll
  for (int mt = 0; mt < 2; ++mt)
#pragma unroll
    for (int nt = 0; nt < 4; ++nt) acc[mt][nt] = zero;

  gemm_mainloop(Ag, Bg, A_s, B_s, acc, t, lane, wm, wn, ln16);

#pragma unroll
  for (int mt = 0; mt < 2; ++mt) {
#pragma unroll
    for (int nt = 0; nt < 4; ++nt) {
      const int e = n0 + nt * 16 + ln16;
      const float be = bO[e];
#pragma unroll
      for (int r = 0; r < 8; ++r) {
        const int s = m0 + mt * 16 + r + 8 * grp;
        const size_t o = ((size_t)(b * SEQ + s)) * DM + e;
        out[o] = x[o] + be + acc[mt][nt][r];
      }
    }
  }
}

// ---------------------------------------------------------------------------
extern "C" void kernel_launch(void* const* d_in, const int* in_sizes, int n_in,
                              void* d_out, int out_size, void* d_ws,
                              size_t ws_size, hipStream_t stream) {
  const float* x            = (const float*)d_in[0];
  const int* st             = (const int*)d_in[1];
  const unsigned char* pm   = (const unsigned char*)d_in[2];
  const float* Wq           = (const float*)d_in[3];
  const float* Wk           = (const float*)d_in[4];
  const float* Wv           = (const float*)d_in[5];
  const float* Wo           = (const float*)d_in[6];
  const float* bO           = (const float*)d_in[7];
  const float* etb          = (const float*)d_in[8];
  const float* srb          = (const float*)d_in[9];
  const float* rsb          = (const float*)d_in[10];
  float* out                = (float*)d_out;

  char* ws = (char*)d_ws;
  _Float16* xh  = (_Float16*)ws;                       // 33,554,432 B
  _Float16* wqh = (_Float16*)(ws + (size_t)33554432);  // 4 x 2 MB weights
  _Float16* wkh = wqh + 1048576;
  _Float16* wvh = wkh + 1048576;
  _Float16* woh = wvh + 1048576;
  _Float16* Qh  = woh + 1048576;                       // 3 x 32 MB QKV
  _Float16* Kh  = Qh + 16777216;
  _Float16* Vh  = Kh + 16777216;
  _Float16* ctx = Vh + 16777216;                       // 32 MB context

  ln_kernel<<<dim3(BDIM * SEQ), 256, 0, stream>>>(x, xh);
  cvt_kernel<<<dim3(4096), 256, 0, stream>>>(Wq, wqh);
  cvt_kernel<<<dim3(4096), 256, 0, stream>>>(Wk, wkh);
  cvt_kernel<<<dim3(4096), 256, 0, stream>>>(Wv, wvh);
  cvt_kernel<<<dim3(4096), 256, 0, stream>>>(Wo, woh);
  gemm_qkv_kernel<<<dim3(16, 3, BDIM), 256, 0, stream>>>(xh, wqh, wkh, wvh, Qh,
                                                         Kh, Vh);
  attn_kernel<<<dim3(NH, BDIM), 256, 0, stream>>>(Qh, Kh, Vh, st, pm, etb, srb,
                                                  rsb, ctx);
  gemm_out_kernel<<<dim3(16, 1, BDIM), 256, 0, stream>>>(ctx, woh, x, bO, out);

  (void)in_sizes; (void)n_in; (void)out_size; (void)ws_size;
}
